// EncoderPlanarLSTM_90477781057685
// MI455X (gfx1250) — matrix operand order repaired and checked by reading, not compiled, not run on hardware
//
#include <hip/hip_runtime.h>
#include <hip/hip_bf16.h>

// ---------------------------------------------------------------------------
// EncoderPlanarLSTM for MI455X (gfx1250): bf16 WMMA recurrence, cluster-split H,
// TDM (tensor_load_to_lds) weight staging, native v_tanh_f32 epilogue.
// ---------------------------------------------------------------------------

typedef __bf16 bf16_t;
typedef __attribute__((ext_vector_type(16))) __bf16 v16bf;
typedef __attribute__((ext_vector_type(8)))  __bf16 v8bf;
typedef __attribute__((ext_vector_type(8)))  float  v8f;
typedef __attribute__((ext_vector_type(4)))  unsigned int u32x4;
typedef __attribute__((ext_vector_type(8)))  unsigned int u32x8;

#define T_STEPS 256
#define BATCH   1024
#define DXC     48
#define DAC     16
#define DIN     64            // I = DX + DA
#define HID     512
#define ZDIM    32
#define FDIM    4
#define KTOT    (DIN + HID)   // 576
#define KC_N    (KTOT / 32)   // 18 K-chunks of 32
#define N4H     (4 * HID)     // 2048 gate columns
#define NT_TOT  (N4H / 16)    // 128 N-tiles of 16

#define CLU_WGS 16            // workgroups per cluster (max)
#define NJ      32            // hidden columns owned per WG
#define NSLICE  (4 * NJ)      // 128 gate columns per WG
#define BT      16            // batch rows per cluster
#define NCLU    (BATCH / BT)  // 64 clusters
#define NWG     (NCLU * CLU_WGS) // 1024 workgroups

#define TILE_BYTES   (KC_N * 1024)          // 18432 B per N-tile (18 x 1KB blocks)
#define ROW_BYTES    (2 * TILE_BYTES)       // 36864 B: WG's 2 tiles per gate
#define ROWSTRIDE_B  (32 * TILE_BYTES)      // 589824 B between gates in Wsw

// LDS: weights 8 tiles * 18 chunks * 1KB = 147456 B + gate staging 8192 B
#define LDS_W_BYTES   (8 * KC_N * 1024)
#define LDS_G_BYTES   (BT * NSLICE * 4)
#define LDS_TOTAL     (LDS_W_BYTES + LDS_G_BYTES)

#if defined(__has_attribute)
#if __has_attribute(amdgpu_cluster_dims)
#define CLUSTER_ATTR __attribute__((amdgpu_cluster_dims(CLU_WGS, 1, 1)))
#endif
#endif
#ifndef CLUSTER_ATTR
#define CLUSTER_ATTR
#endif

// ------------------------- workspace layout (bytes) -------------------------
#define OFF_W     0                                   // swizzled weights bf16
#define OFF_BIAS  (OFF_W + N4H * KTOT * 2)            // f32 [2048]
#define OFF_Y     (OFF_BIAS + N4H * 4)                // bf16 [T][B][64]
#define OFF_HEX   (OFF_Y + (size_t)T_STEPS * BATCH * DIN * 2)  // bf16 [2][B][512]
#define OFF_HFIN  (OFF_HEX + (size_t)2 * BATCH * HID * 2)      // f32 [B][512]

// --------------------------- device helpers --------------------------------
__device__ __forceinline__ float fast_tanh(float x) {
  float r;
  // v_tanh_f32 is a gfx1250 TRANS op; trailing v_nop honors the 1-instruction
  // TRANS hazard window (ISA 7.4 #3) since the compiler can't see inside asm.
  asm volatile("v_tanh_f32 %0, %1\n\tv_nop" : "=v"(r) : "v"(x));
  return r;
}
__device__ __forceinline__ float fast_sigmoid(float x) {
  return __builtin_amdgcn_rcpf(1.0f + __expf(-x));
}

// ---------------------------------------------------------------------------
// Prep 1: swizzle weights into WMMA B-layout blocks, bf16; also combined bias.
// Block layout: [nt(128)][kc(18)][lane(32)][16 bf16] where
//   n = nt*16 + (lane&15), k = kc*32 + (lane>>4)*16 + j
// ---------------------------------------------------------------------------
__global__ void prep_weights_kernel(const float* __restrict__ w_ih,
                                    const float* __restrict__ w_hh,
                                    const float* __restrict__ b_ih,
                                    const float* __restrict__ b_hh,
                                    bf16_t* __restrict__ Wsw,
                                    float* __restrict__ bias) {
  int f = blockIdx.x * blockDim.x + threadIdx.x;
  const int total = NT_TOT * KC_N * 32 * 16;  // 1,179,648
  if (f < total) {
    int j    = f & 15;
    int lane = (f >> 4) & 31;
    int blk  = f >> 9;             // nt*KC_N + kc
    int kc   = blk % KC_N;
    int nt   = blk / KC_N;
    int n = nt * 16 + (lane & 15);
    int k = kc * 32 + (lane >> 4) * 16 + j;
    float v = (k < DIN) ? w_ih[n * DIN + k] : w_hh[n * HID + (k - DIN)];
    Wsw[f] = (bf16_t)v;
  }
  if (f < N4H) bias[f] = b_ih[f] + b_hh[f];
}

// ---------------------------------------------------------------------------
// Prep 2: masked, concatenated, time-reversed input in bf16.
// ---------------------------------------------------------------------------
__global__ void prep_input_kernel(const float* __restrict__ x,
                                  const float* __restrict__ a,
                                  const float* __restrict__ mask,
                                  bf16_t* __restrict__ ybf) {
  size_t gid = (size_t)blockIdx.x * blockDim.x + threadIdx.x;
  if (gid >= (size_t)T_STEPS * BATCH * DIN) return;
  int i = (int)(gid & (DIN - 1));
  size_t tb = gid >> 6;
  int b = (int)(tb % BATCH);
  int t = (int)(tb / BATCH);
  int ts = T_STEPS - 1 - t;
  float v;
  if (i < DXC) {
    size_t s = ((size_t)ts * BATCH + b) * DXC + i;
    v = x[s] * mask[s];
  } else {
    v = a[((size_t)ts * BATCH + b) * DAC + (i - DXC)];
  }
  ybf[gid] = (bf16_t)v;
}

// ---------------------------------------------------------------------------
// Prep 3: zero h-exchange buffer 0 (read at t==0).
// ---------------------------------------------------------------------------
__global__ void prep_hzero_kernel(unsigned int* __restrict__ hex0) {
  int gid = blockIdx.x * blockDim.x + threadIdx.x;
  if (gid < BATCH * HID * 2 / 4) hex0[gid] = 0u;
}

// ---------------------------------------------------------------------------
// Main LSTM kernel: 64 clusters x 16 WGs, 256 threads (8 waves) each.
// ---------------------------------------------------------------------------
CLUSTER_ATTR __launch_bounds__(256)
__global__ void lstm_main_kernel(const bf16_t* __restrict__ Wsw,
                                 const float* __restrict__ bias,
                                 const bf16_t* __restrict__ ybf,
                                 bf16_t* __restrict__ hex,
                                 float* __restrict__ hfin) {
  extern __shared__ char smem[];
  bf16_t* w_lds = (bf16_t*)smem;
  float*  gbuf  = (float*)(smem + LDS_W_BYTES);

  const int tid  = threadIdx.x;
  const int lane = tid & 31;
  const int wv   = tid >> 5;               // 0..7
  const int g    = blockIdx.x & (CLU_WGS - 1);   // workgroup-in-cluster
  const int cl   = blockIdx.x >> 4;              // cluster index
  const int q    = wv >> 1;                // gate 0..3 (i,f,g,o)
  const int tt   = wv & 1;                 // 16-col sub-tile within gate slice

  // ---- Stage this WG's weight slice via the Tensor Data Mover:
  // 2D tile: 4 rows (one per gate) of 36864 B, global row stride 589824 B,
  // landing contiguously in LDS. D# fields per ISA ch.8; data_size = 8 bytes.
  if (tid < 32) {
    unsigned lds_base = (unsigned)(uintptr_t)(void*)w_lds;   // LDS aperture low 32b
    unsigned long long ga =
        (unsigned long long)(uintptr_t)(const void*)Wsw +
        (unsigned long long)g * (unsigned long long)ROW_BYTES;
    u32x4 d0;
    d0[0] = 1u;                                   // count=1, user descriptor
    d0[1] = lds_base;                             // lds_addr
    d0[2] = (unsigned)(ga & 0xFFFFFFFFull);       // global_addr[31:0]
    d0[3] = (unsigned)((ga >> 32) & 0x01FFFFFFull) | (2u << 30); // addr hi | type=2
    const unsigned DIM0   = ROWSTRIDE_B / 8u;     // 73728: tensor dim0 (units of 8B)
    const unsigned TILE0  = ROW_BYTES / 8u;       // 4608
    u32x8 d1;
    d1[0] = (3u << 16);                           // data_size = 3 (8 bytes)
    d1[1] = (DIM0 & 0xFFFFu) << 16;               // tensor_dim0[15:0]
    d1[2] = (DIM0 >> 16) | (4u << 16);            // tensor_dim0[31:16] | tensor_dim1 lo
    d1[3] = (TILE0 << 16);                        // tensor_dim1 hi=0 | tile_dim0
    d1[4] = 4u;                                   // tile_dim1=4, tile_dim2=0
    d1[5] = DIM0;                                 // tensor_dim0_stride[31:0]
    d1[6] = 0u;                                   // stride hi | dim1_stride lo
    d1[7] = 0u;
    asm volatile("tensor_load_to_lds %0, %1" :: "s"(d0), "s"(d1) : "memory");
    __builtin_amdgcn_s_wait_tensorcnt(0);
  }
  __syncthreads();

  // Per-lane invariants for the WMMA operand layouts (ISA 7.12.2):
  const int m     = lane & 15;                   // A row (batch row in tile)
  const int b_row = cl * BT + m;                 // global batch row
  const int kb    = (lane >> 4) * 16;            // A: one 16-elem run at kb
  const int nglob = q * HID + g * NJ + tt * 16 + (lane & 15);
  const float bv  = bias[nglob];                 // fold bias into acc init
  const int tl_w  = wv;                          // wave's LDS tile index

  // c-state lives in registers: thread owns items tid and tid+256.
  const int mm0 = tid >> 5, jj0 = tid & 31;      // item 0: (mm0, jj0)
  const int mm1 = mm0 + 8;                       // item 1: (mm0+8, jj0)
  float c0 = 0.0f, c1 = 0.0f, h0 = 0.0f, h1 = 0.0f;

  for (int t = 0; t < T_STEPS; ++t) {
    const bf16_t* hcur = hex + (size_t)(t & 1) * BATCH * HID;
    bf16_t*       hnxt = hex + (size_t)((t & 1) ^ 1) * BATCH * HID;

    // ---- Load A fragments: concat(y_t row, h_prev row), 18 chunks of K=32
    v16bf aF[KC_N];
    {
      const bf16_t* yrow = ybf + ((size_t)t * BATCH + b_row) * DIN;
      const bf16_t* hrow = hcur + (size_t)b_row * HID;
#pragma unroll
      for (int kc = 0; kc < KC_N; ++kc) {
        const bf16_t* base = (kc < 2) ? (yrow + kc * 32) : (hrow + (kc - 2) * 32);
        v8bf lo = *(const v8bf*)(base + kb);
        v8bf hi = *(const v8bf*)(base + kb + 8);
        v16bf av;
#pragma unroll
        for (int j = 0; j < 8; ++j) { av[j] = lo[j]; av[8 + j] = hi[j]; }
        aF[kc] = av;
      }
    }
    {
      int tp = (t + 1) & (T_STEPS - 1);          // wraps; no loop-splitting branch
      if (lane == 0)
        __builtin_prefetch(ybf + ((size_t)tp * BATCH + cl * BT) * DIN, 0, 1);
    }

    // ---- 18 WMMA ops: acc(16x16 f32) += A(16x32 bf16) x B(32x16 bf16)
    v8f acc;
#pragma unroll
    for (int r = 0; r < 8; ++r) acc[r] = bv;
#pragma unroll
    for (int kc = 0; kc < KC_N; ++kc) {
      v16bf bF = *(const v16bf*)(w_lds + (size_t)(tl_w * KC_N + kc) * 512 + lane * 16);
      acc = __builtin_amdgcn_wmma_f32_16x16x32_bf16(
          /*neg_a=*/false, aF[kc], /*neg_b=*/false, bF,
          /*c_mod=*/(short)0, acc, /*reuse_a=*/false, /*reuse_b=*/false);
    }

    // ---- Scatter D tile to gate staging LDS (C/D layout: m = r + 8*(lane>>4))
    const int lnb = q * NJ + tt * 16 + (lane & 15);
    const int mh  = (lane >> 4) * 8;
#pragma unroll
    for (int r = 0; r < 8; ++r) gbuf[(mh + r) * NSLICE + lnb] = acc[r];
    __syncthreads();

    // ---- Elementwise gate math + state update (2 items per thread, regs for c)
    {
      float gi = gbuf[mm0 * NSLICE + 0 * NJ + jj0];
      float gf = gbuf[mm0 * NSLICE + 1 * NJ + jj0];
      float gg = gbuf[mm0 * NSLICE + 2 * NJ + jj0];
      float go = gbuf[mm0 * NSLICE + 3 * NJ + jj0];
      c0 = fast_sigmoid(gf) * c0 + fast_sigmoid(gi) * fast_tanh(gg);
      h0 = fast_sigmoid(go) * fast_tanh(c0);
      hnxt[(size_t)(cl * BT + mm0) * HID + g * NJ + jj0] = (bf16_t)h0;
    }
    {
      float gi = gbuf[mm1 * NSLICE + 0 * NJ + jj0];
      float gf = gbuf[mm1 * NSLICE + 1 * NJ + jj0];
      float gg = gbuf[mm1 * NSLICE + 2 * NJ + jj0];
      float go = gbuf[mm1 * NSLICE + 3 * NJ + jj0];
      c1 = fast_sigmoid(gf) * c1 + fast_sigmoid(gi) * fast_tanh(gg);
      h1 = fast_sigmoid(go) * fast_tanh(c1);
      hnxt[(size_t)(cl * BT + mm1) * HID + g * NJ + jj0] = (bf16_t)h1;
    }

    // ---- Make h slice visible to cluster peers, then cluster-wide barrier
    __threadfence();
    __builtin_amdgcn_s_cluster_barrier();
    __syncthreads();
  }

  // ---- Final hidden state (f32) for the heads
  hfin[(size_t)(cl * BT + mm0) * HID + g * NJ + jj0] = h0;
  hfin[(size_t)(cl * BT + mm1) * HID + g * NJ + jj0] = h1;
}

// ---------------------------------------------------------------------------
// Heads: mu = exp(h@lin^T+b)/10 ; log_var = h@lv^T+b - 5 ; u, w, b plain.
// d_out = [mu | log_var | u | w | b] flattened, f32.
// ---------------------------------------------------------------------------
__global__ void heads_kernel(const float* __restrict__ h,
                             const float* __restrict__ lin_w, const float* __restrict__ lin_b,
                             const float* __restrict__ lv_w,  const float* __restrict__ lv_b,
                             const float* __restrict__ u_w,   const float* __restrict__ u_b,
                             const float* __restrict__ ww_w,  const float* __restrict__ ww_b,
                             const float* __restrict__ bf_w,  const float* __restrict__ bf_b,
                             float* __restrict__ out) {
  const int NMU = BATCH * ZDIM;          // 32768
  const int NU  = BATCH * FDIM * ZDIM;   // 131072
  const int NB  = BATCH * FDIM;          // 4096
  const int total = 2 * NMU + 2 * NU + NB;
  int gid = blockIdx.x * blockDim.x + threadIdx.x;
  if (gid >= total) return;

  int idx = gid, b, mode;
  const float* wrow; float bb;
  if (idx < NMU) {
    b = idx / ZDIM; int z = idx % ZDIM; wrow = lin_w + z * HID; bb = lin_b[z]; mode = 0;
  } else if (idx < 2 * NMU) {
    idx -= NMU; b = idx / ZDIM; int z = idx % ZDIM; wrow = lv_w + z * HID; bb = lv_b[z]; mode = 1;
  } else if (idx < 2 * NMU + NU) {
    idx -= 2 * NMU; b = idx / (FDIM * ZDIM); int r = idx % (FDIM * ZDIM);
    wrow = u_w + r * HID; bb = u_b[r]; mode = 2;
  } else if (idx < 2 * NMU + 2 * NU) {
    idx -= 2 * NMU + NU; b = idx / (FDIM * ZDIM); int r = idx % (FDIM * ZDIM);
    wrow = ww_w + r * HID; bb = ww_b[r]; mode = 2;
  } else {
    idx -= 2 * NMU + 2 * NU; b = idx / FDIM; int r = idx % FDIM;
    wrow = bf_w + r * HID; bb = bf_b[r]; mode = 2;
  }

  const float* hrow = h + (size_t)b * HID;
  float acc = bb;
#pragma unroll 4
  for (int k = 0; k < HID; k += 4) {
    float4 hv = *(const float4*)(hrow + k);
    float4 wv = *(const float4*)(wrow + k);
    acc += hv.x * wv.x + hv.y * wv.y + hv.z * wv.z + hv.w * wv.w;
  }
  if (mode == 0)      acc = __expf(acc) / 10.0f;
  else if (mode == 1) acc -= 5.0f;
  out[gid] = acc;
}

// ---------------------------------------------------------------------------
extern "C" void kernel_launch(void* const* d_in, const int* in_sizes, int n_in,
                              void* d_out, int out_size, void* d_ws, size_t ws_size,
                              hipStream_t stream) {
  const float* x     = (const float*)d_in[0];
  const float* a     = (const float*)d_in[1];
  const float* mask  = (const float*)d_in[2];
  const float* w_ih  = (const float*)d_in[3];
  const float* w_hh  = (const float*)d_in[4];
  const float* b_ih  = (const float*)d_in[5];
  const float* b_hh  = (const float*)d_in[6];
  const float* lin_w = (const float*)d_in[7];
  const float* lin_b = (const float*)d_in[8];
  const float* lv_w  = (const float*)d_in[9];
  const float* lv_b  = (const float*)d_in[10];
  const float* u_w   = (const float*)d_in[11];
  const float* u_b   = (const float*)d_in[12];
  const float* ww_w  = (const float*)d_in[13];
  const float* ww_b  = (const float*)d_in[14];
  const float* bf_w  = (const float*)d_in[15];
  const float* bf_b  = (const float*)d_in[16];

  char* ws = (char*)d_ws;
  bf16_t* Wsw  = (bf16_t*)(ws + OFF_W);
  float*  bias = (float*)(ws + OFF_BIAS);
  bf16_t* ybf  = (bf16_t*)(ws + OFF_Y);
  bf16_t* hex  = (bf16_t*)(ws + OFF_HEX);
  float*  hfin = (float*)(ws + OFF_HFIN);

  {
    int total = NT_TOT * KC_N * 32 * 16;
    prep_weights_kernel<<<(total + 255) / 256, 256, 0, stream>>>(
        w_ih, w_hh, b_ih, b_hh, Wsw, bias);
  }
  {
    size_t total = (size_t)T_STEPS * BATCH * DIN;
    prep_input_kernel<<<(unsigned)((total + 255) / 256), 256, 0, stream>>>(
        x, a, mask, ybf);
  }
  {
    int total = BATCH * HID * 2 / 4;
    prep_hzero_kernel<<<(total + 255) / 256, 256, 0, stream>>>((unsigned int*)hex);
  }
  lstm_main_kernel<<<NWG, 256, LDS_TOTAL, stream>>>(Wsw, bias, ybf, hex, hfin);
  {
    int total = 2 * BATCH * ZDIM + 2 * BATCH * FDIM * ZDIM + BATCH * FDIM;
    heads_kernel<<<(total + 255) / 256, 256, 0, stream>>>(
        hfin, lin_w, lin_b, lv_w, lv_b, u_w, u_b, ww_w, ww_b, bf_w, bf_b,
        (float*)d_out);
  }
}
